// Attention_61357902791236
// MI455X (gfx1250) — matrix operand order repaired and checked by reading, not compiled, not run on hardware
//
#include <hip/hip_runtime.h>
#include <hip/hip_bf16.h>
#include <stdint.h>

// ---- problem constants ----
#define NWIN   1024          // 4 batches * 16*16 windows
#define NTOK   65            // 1 global token + 8*8 window tokens
#define ATT_SCALE 0.17677669529663689f   // 32^-0.5
#define MROWS  (NWIN*NTOK)   // 66560 rows, exactly 4160 * 16

// d_out regions (floats)
#define XO_OFF   0u
#define GT_OFF   16777216u   // 4*16384*256
#define SKIP_OFF 17039360u   // + 1024*256

typedef __attribute__((ext_vector_type(16))) __bf16 v16bf;
typedef __attribute__((ext_vector_type(8)))  float  v8f;

union Frag16 { v16bf v; uint4 q[2]; };

__device__ inline unsigned short f2bf(float f) {
  unsigned int u = __float_as_uint(f);
  u += 0x7FFFu + ((u >> 16) & 1u);       // round-to-nearest-even
  return (unsigned short)(u >> 16);
}
__device__ inline v8f v8f_zero() { v8f z = {0.f,0.f,0.f,0.f,0.f,0.f,0.f,0.f}; return z; }

// ---------------------------------------------------------------------------
// Weight packing: fp32 -> bf16, (N,K) row-major for WMMA B-fragments.
// wsr[oc][t*256+ic] = sr_w[oc][ic][dy][dx], t = dy*2+dx   (K = 1024)
// ---------------------------------------------------------------------------
__global__ __launch_bounds__(256) void pack_w_k(
    const float* __restrict__ qw, const float* __restrict__ kvw,
    const float* __restrict__ pw, const float* __restrict__ srw,
    unsigned short* wq, unsigned short* wkv,
    unsigned short* wp, unsigned short* wsr) {
  int idx = blockIdx.x * 256 + threadIdx.x;
  if (idx < 65536) {
    wq[idx] = f2bf(qw[idx]);
  } else if (idx < 65536 + 131072) {
    int j = idx - 65536; wkv[j] = f2bf(kvw[j]);
  } else if (idx < 65536 + 131072 + 65536) {
    int j = idx - 196608; wp[j] = f2bf(pw[j]);
  } else {
    int j = idx - 262144;
    int oc = j >> 10, rem = j & 1023, t = rem >> 8, ic = rem & 255;
    int dy = t >> 1, dx = t & 1;
    wsr[j] = f2bf(srw[(((size_t)oc * 256 + ic) * 2 + dy) * 2 + dx]);
  }
}

// ---------------------------------------------------------------------------
// Build xc (windowed tokens + global token, bf16), seed xkv token 0 with g,
// and emit the "skip" output (fp32 copy of gt).
// ---------------------------------------------------------------------------
__global__ __launch_bounds__(256) void build_inputs_k(
    const float* __restrict__ x, const float* __restrict__ gt,
    unsigned short* __restrict__ xc,
    unsigned short* __restrict__ xkv,
    float* __restrict__ dout) {
  int blk = blockIdx.x;                 // win*65 + tok
  int win = blk / NTOK, tok = blk - win * NTOK;
  int c = threadIdx.x;
  if (tok == 0) {
    float g = gt[(size_t)win * 256 + c];
    unsigned short gb = f2bf(g);
    xc[(size_t)blk * 256 + c]  = gb;
    xkv[(size_t)blk * 256 + c] = gb;
    dout[SKIP_OFF + (size_t)win * 256 + c] = g;
  } else {
    int p = tok - 1, py = p >> 3, px = p & 7;
    int b = win >> 8, wi = win & 255, wh = wi >> 4, ww = wi & 15;
    int y = wh * 8 + py, xcol = ww * 8 + px;
    float v = x[((size_t)(b * 16384) + y * 128 + xcol) * 256 + c];
    xc[(size_t)blk * 256 + c] = f2bf(v);
  }
}

// ---------------------------------------------------------------------------
// Zero helper (for V^T padding).
// ---------------------------------------------------------------------------
__global__ __launch_bounds__(256) void zero_k(uint4* p, unsigned long long n16) {
  unsigned long long i = (unsigned long long)blockIdx.x * blockDim.x + threadIdx.x;
  if (i < n16) { uint4 z = make_uint4(0u,0u,0u,0u); p[i] = z; }
}

// ---------------------------------------------------------------------------
// Bilinear align-corners upsample 8x8 -> 16x16 per window, written as the
// conv tap tensor up[win][opix][tap][ic] (bf16), K layout for conv GEMM.
// ---------------------------------------------------------------------------
__global__ __launch_bounds__(256) void bilinear_k(
    const float* __restrict__ x, unsigned short* __restrict__ up) {
  int idx = blockIdx.x * 256 + threadIdx.x;
  int c = idx & 255, rest = idx >> 8;
  int u = rest & 255, win = rest >> 8;
  int uy = u >> 4, ux = u & 15;
  float posy = uy * (7.0f / 15.0f);
  int y0 = (int)posy; if (y0 > 6) y0 = 6;
  float ty = posy - (float)y0;
  float posx = ux * (7.0f / 15.0f);
  int x0 = (int)posx; if (x0 > 6) x0 = 6;
  float tx = posx - (float)x0;
  int b = win >> 8, wi = win & 255, wh = wi >> 4, ww = wi & 15;
  const float* base = x + ((size_t)(b * 16384)) * 256 + c;
  #define LD(py,px) base[((size_t)((wh*8+(py))*128 + ww*8+(px))) * 256]
  float v00 = LD(y0, x0),   v01 = LD(y0, x0+1);
  float v10 = LD(y0+1, x0), v11 = LD(y0+1, x0+1);
  #undef LD
  float top = v00 * (1.f - tx) + v01 * tx;
  float bot = v10 * (1.f - tx) + v11 * tx;
  float val = top * (1.f - ty) + bot * ty;
  int opix = (uy >> 1) * 8 + (ux >> 1);
  int t = (uy & 1) * 2 + (ux & 1);
  up[((size_t)(win * 64 + opix) * 4 + t) * 256 + c] = f2bf(val);
}

// ---------------------------------------------------------------------------
// Generic bf16 WMMA GEMM (K fixed at 256): C[M,N] = A[M,256] @ W[N,256]^T +
// bias.  A tile (16x256, 8 KB) staged in LDS once per block; B fragments
// preloaded per k-step so loads clause together ahead of the wmma burst.
// MODE 0: store bf16 (q proj). MODE 1: k rows / transposed-padded v (kv).
// MODE 2: fp32 scatter to final outputs (proj).
// ---------------------------------------------------------------------------
template<int TPW, int MODE>
__global__ __launch_bounds__(256) void gemm_bias_k(
    const unsigned short* __restrict__ A,
    const unsigned short* __restrict__ W,
    const float* __restrict__ bias,
    int N,
    unsigned short* __restrict__ outb,
    unsigned short* __restrict__ kbuf,
    unsigned short* __restrict__ vt,
    float* __restrict__ dout) {
  __shared__ __align__(16) unsigned short aTile[16][256];   // 8 KB
  const int lane = threadIdx.x & 31;
  const int wave = threadIdx.x >> 5;
  const int r = lane & 15, hf = lane >> 4;
  const int rowBase = blockIdx.x * 16;

  // cooperative A-tile load: thread t copies 16 elems (row t>>4, seg t&15)
  {
    int row = threadIdx.x >> 4, seg = threadIdx.x & 15;
    const uint4* src = (const uint4*)(A + ((size_t)(rowBase + row)) * 256 + seg * 16);
    *(uint4*)&aTile[row][seg * 16]     = src[0];
    *(uint4*)&aTile[row][seg * 16 + 8] = src[1];
  }
  __syncthreads();

  v8f acc[TPW];
  #pragma unroll
  for (int t = 0; t < TPW; ++t) acc[t] = v8f_zero();

  #pragma unroll 1
  for (int ks = 0; ks < 8; ++ks) {
    Frag16 a;
    a.q[0] = *(const uint4*)&aTile[r][ks * 32 + hf * 16];
    a.q[1] = *(const uint4*)&aTile[r][ks * 32 + hf * 16 + 8];
    Frag16 b[TPW];
    #pragma unroll
    for (int t = 0; t < TPW; ++t) {
      int n = (wave * TPW + t) * 16 + r;
      const unsigned short* bb = W + (size_t)n * 256 + ks * 32 + hf * 16;
      b[t].q[0] = *(const uint4*)(bb);
      b[t].q[1] = *(const uint4*)(bb + 8);
    }
    #pragma unroll
    for (int t = 0; t < TPW; ++t)
      acc[t] = __builtin_amdgcn_wmma_f32_16x16x32_bf16(
          false, a.v, false, b[t].v, (short)0, acc[t], false, false);
  }

  #pragma unroll
  for (int t = 0; t < TPW; ++t) {
    int n = (wave * TPW + t) * 16 + r;
    float bv = bias[n];
    #pragma unroll
    for (int v = 0; v < 8; ++v) {
      int row = rowBase + v + 8 * hf;
      float val = acc[t][v] + bv;
      if (MODE == 0) {
        outb[(size_t)row * N + n] = f2bf(val);
      } else if (MODE == 1) {
        int win = row / NTOK, tok = row - win * NTOK;
        if (n < 256) {
          kbuf[(size_t)row * 256 + n] = f2bf(val);
        } else {
          int c2 = n - 256, hd = c2 >> 5, d = c2 & 31;
          vt[((size_t)(win * 8 + hd) * 32 + d) * 96 + tok] = f2bf(val);
        }
      } else {
        int win = row / NTOK, tok = row - win * NTOK;
        if (tok == 0) {
          dout[GT_OFF + (size_t)win * 256 + n] = val;
        } else {
          int p = tok - 1, py = p >> 3, px = p & 7;
          int b2 = win >> 8, wi = win & 255, wh = wi >> 4, ww = wi & 15;
          int y = wh * 8 + py, xcol = ww * 8 + px;
          dout[XO_OFF + ((size_t)(b2 * 16384) + y * 128 + xcol) * 256 + n] = val;
        }
      }
    }
  }
}

// ---------------------------------------------------------------------------
// Conv-as-GEMM (64x1024 @ 1024x256 per window) + bias + fused LayerNorm.
// 8 waves/block: wave = (colhalf ch, M-tile mt).  Each wave owns 16 pixels x
// 128 cols = 8 acc tiles (64 VGPRs) -> no spills.  LN combines the two
// column halves through a small LDS array.
// ---------------------------------------------------------------------------
__global__ __launch_bounds__(256) void conv_ln_k(
    const unsigned short* __restrict__ up,
    const unsigned short* __restrict__ wsr,
    const float* __restrict__ srb,
    const float* __restrict__ nw, const float* __restrict__ nb,
    unsigned short* __restrict__ xkv) {
  __shared__ float red1[2][64];
  __shared__ float red2[2][64];
  const int lane = threadIdx.x & 31;
  const int wave = threadIdx.x >> 5;   // 0..7
  const int mt = wave & 3;             // M tile (16 pixels)
  const int ch = wave >> 2;            // column half (128 cols)
  const int r = lane & 15, hf = lane >> 4;
  const int win = blockIdx.x;

  v8f acc[8];
  #pragma unroll
  for (int t = 0; t < 8; ++t) acc[t] = v8f_zero();

  const unsigned short* arow  = up + ((size_t)(win * 64) + mt * 16 + r) * 1024;
  const unsigned short* wbase = wsr + (size_t)(ch * 128 + r) * 1024;
  #pragma unroll 1
  for (int ks = 0; ks < 32; ++ks) {
    Frag16 a;
    a.q[0] = *(const uint4*)(arow + ks * 32 + hf * 16);
    a.q[1] = *(const uint4*)(arow + ks * 32 + hf * 16 + 8);
    Frag16 b[8];
    #pragma unroll
    for (int nt = 0; nt < 8; ++nt) {
      const unsigned short* bb = wbase + (size_t)nt * 16 * 1024 + ks * 32 + hf * 16;
      b[nt].q[0] = *(const uint4*)bb;
      b[nt].q[1] = *(const uint4*)(bb + 8);
    }
    #pragma unroll
    for (int nt = 0; nt < 8; ++nt)
      acc[nt] = __builtin_amdgcn_wmma_f32_16x16x32_bf16(
          false, a.v, false, b[nt].v, (short)0, acc[nt], false, false);
  }
  // conv bias
  #pragma unroll
  for (int nt = 0; nt < 8; ++nt) {
    float bv = srb[ch * 128 + nt * 16 + r];
    #pragma unroll
    for (int v = 0; v < 8; ++v) acc[nt][v] += bv;
  }
  // partial LN sums over this wave's 128 cols, per pixel row
  float s1v[8], s2v[8];
  #pragma unroll
  for (int v = 0; v < 8; ++v) {
    float a1 = 0.f, a2 = 0.f;
    #pragma unroll
    for (int nt = 0; nt < 8; ++nt) { float xv = acc[nt][v]; a1 += xv; a2 += xv * xv; }
    for (int off = 1; off < 16; off <<= 1) {
      a1 += __shfl_xor(a1, off, 32);
      a2 += __shfl_xor(a2, off, 32);
    }
    s1v[v] = a1; s2v[v] = a2;
  }
  if (r == 0) {            // lanes 0 and 16: rows v+0 / v+8 of this M tile
    #pragma unroll
    for (int v = 0; v < 8; ++v) {
      int pix = mt * 16 + v + 8 * hf;
      red1[ch][pix] = s1v[v];
      red2[ch][pix] = s2v[v];
    }
  }
  __syncthreads();
  float mean[8], rstd[8];
  #pragma unroll
  for (int v = 0; v < 8; ++v) {
    int pix = mt * 16 + v + 8 * hf;
    float t1 = red1[0][pix] + red1[1][pix];
    float t2 = red2[0][pix] + red2[1][pix];
    float mu = t1 * (1.0f / 256.0f);
    float var = t2 * (1.0f / 256.0f) - mu * mu;
    mean[v] = mu;
    rstd[v] = rsqrtf(var + 1e-5f);
  }
  #pragma unroll
  for (int nt = 0; nt < 8; ++nt) {
    int col = ch * 128 + nt * 16 + r;
    float w = nw[col], b2 = nb[col];
    #pragma unroll
    for (int v = 0; v < 8; ++v) {
      int pix = mt * 16 + v + 8 * hf;
      float val = (acc[nt][v] - mean[v]) * rstd[v] * w + b2;
      xkv[((size_t)(win * NTOK) + 1 + pix) * 256 + col] = f2bf(val);
    }
  }
}

// ---------------------------------------------------------------------------
// Attention: one block per window, one wave per head. n=65 padded to 80.
// Scores via WMMA, masked softmax in registers, P staged through per-wave
// LDS (row stride 96, cols 80..95 zero), PV via WMMA on pre-transposed V.
// ---------------------------------------------------------------------------
__global__ __launch_bounds__(256) void attention_k(
    const unsigned short* __restrict__ qb,
    const unsigned short* __restrict__ kb,
    const unsigned short* __restrict__ vt,
    unsigned short* __restrict__ ob) {
  __shared__ __align__(16) unsigned short pLds[8][16][96];
  const int lane = threadIdx.x & 31;
  const int h = threadIdx.x >> 5;       // head
  const int r = lane & 15, hf = lane >> 4;
  const int win = blockIdx.x;

  // zero padding columns 80..95 (wave-local region)
  for (int i = lane; i < 256; i += 32) pLds[h][i >> 4][80 + (i & 15)] = 0;

  #pragma unroll 1
  for (int mt = 0; mt < 5; ++mt) {
    int tokA = mt * 16 + r;
    Frag16 qa;
    if (tokA < NTOK) {
      const unsigned short* qbase = qb + ((size_t)(win * NTOK) + tokA) * 256 + h * 32;
      qa.q[0] = *(const uint4*)(qbase + hf * 16);
      qa.q[1] = *(const uint4*)(qbase + hf * 16 + 8);
    } else {
      qa.q[0] = make_uint4(0u,0u,0u,0u); qa.q[1] = make_uint4(0u,0u,0u,0u);
    }

    v8f s[5];
    #pragma unroll
    for (int nt = 0; nt < 5; ++nt) {
      int m = nt * 16 + r;
      Frag16 kf;
      if (m < NTOK) {
        const unsigned short* kbase = kb + ((size_t)(win * NTOK) + m) * 256 + h * 32 + hf * 16;
        kf.q[0] = *(const uint4*)kbase;
        kf.q[1] = *(const uint4*)(kbase + 8);
      } else {
        kf.q[0] = make_uint4(0u,0u,0u,0u); kf.q[1] = make_uint4(0u,0u,0u,0u);
      }
      s[nt] = __builtin_amdgcn_wmma_f32_16x16x32_bf16(
          false, qa.v, false, kf.v, (short)0, v8f_zero(), false, false);
    }

    bool vc[5];
    #pragma unroll
    for (int nt = 0; nt < 5; ++nt) vc[nt] = (nt * 16 + r) < NTOK;

    float mrow[8];
    #pragma unroll
    for (int v = 0; v < 8; ++v) mrow[v] = -3.0e38f;
    #pragma unroll
    for (int nt = 0; nt < 5; ++nt) {
      #pragma unroll
      for (int v = 0; v < 8; ++v) {
        float xv = vc[nt] ? s[nt][v] * ATT_SCALE : -3.0e38f;
        mrow[v] = fmaxf(mrow[v], xv);
      }
    }
    #pragma unroll
    for (int v = 0; v < 8; ++v)
      for (int off = 1; off < 16; off <<= 1)
        mrow[v] = fmaxf(mrow[v], __shfl_xor(mrow[v], off, 32));

    float rsum[8];
    #pragma unroll
    for (int v = 0; v < 8; ++v) rsum[v] = 0.f;
    #pragma unroll
    for (int nt = 0; nt < 5; ++nt) {
      #pragma unroll
      for (int v = 0; v < 8; ++v) {
        float e = vc[nt] ? __expf(s[nt][v] * ATT_SCALE - mrow[v]) : 0.f;
        s[nt][v] = e;
        rsum[v] += e;
      }
    }
    #pragma unroll
    for (int v = 0; v < 8; ++v) {
      for (int off = 1; off < 16; off <<= 1) rsum[v] += __shfl_xor(rsum[v], off, 32);
      rsum[v] = 1.0f / rsum[v];
    }
    #pragma unroll
    for (int nt = 0; nt < 5; ++nt) {
      #pragma unroll
      for (int v = 0; v < 8; ++v)
        pLds[h][v + 8 * hf][nt * 16 + r] = f2bf(s[nt][v] * rsum[v]);
    }

    asm volatile("s_wait_dscnt 0" ::: "memory");   // wave-local LDS RAW

    v8f o[2];
    o[0] = v8f_zero(); o[1] = v8f_zero();
    #pragma unroll
    for (int ksx = 0; ksx < 3; ++ksx) {
      Frag16 pa;
      pa.q[0] = *(const uint4*)&pLds[h][r][ksx * 32 + hf * 16];
      pa.q[1] = *(const uint4*)&pLds[h][r][ksx * 32 + hf * 16 + 8];
      #pragma unroll
      for (int n2 = 0; n2 < 2; ++n2) {
        int d = n2 * 16 + r;
        const unsigned short* vb = vt + ((size_t)(win * 8 + h) * 32 + d) * 96 + ksx * 32 + hf * 16;
        Frag16 vf;
        vf.q[0] = *(const uint4*)vb;
        vf.q[1] = *(const uint4*)(vb + 8);
        o[n2] = __builtin_amdgcn_wmma_f32_16x16x32_bf16(
            false, pa.v, false, vf.v, (short)0, o[n2], false, false);
      }
    }
    #pragma unroll
    for (int n2 = 0; n2 < 2; ++n2) {
      #pragma unroll
      for (int v = 0; v < 8; ++v) {
        int tok = mt * 16 + v + 8 * hf;
        if (tok < NTOK)
          ob[((size_t)(win * NTOK) + tok) * 256 + h * 32 + n2 * 16 + r] = f2bf(o[n2][v]);
      }
    }
  }
}

// ---------------------------------------------------------------------------
extern "C" void kernel_launch(void* const* d_in, const int* in_sizes, int n_in,
                              void* d_out, int out_size, void* d_ws, size_t ws_size,
                              hipStream_t stream) {
  (void)in_sizes; (void)n_in; (void)out_size; (void)ws_size;
  const float* x    = (const float*)d_in[0];
  const float* gt   = (const float*)d_in[1];
  const float* q_w  = (const float*)d_in[2];
  const float* q_b  = (const float*)d_in[3];
  const float* kv_w = (const float*)d_in[4];
  const float* kv_b = (const float*)d_in[5];
  const float* sr_w = (const float*)d_in[6];
  const float* sr_b = (const float*)d_in[7];
  const float* nw   = (const float*)d_in[8];
  const float* nb   = (const float*)d_in[9];
  const float* p_w  = (const float*)d_in[10];
  const float* p_b  = (const float*)d_in[11];
  float* out = (float*)d_out;

  // workspace carving (bytes, all 16B aligned)
  char* ws = (char*)d_ws;
  unsigned short* wq   = (unsigned short*)(ws + 0);            // 128 KB
  unsigned short* wkv  = (unsigned short*)(ws + 131072ULL);    // 256 KB
  unsigned short* wp   = (unsigned short*)(ws + 393216ULL);    // 128 KB
  unsigned short* wsr  = (unsigned short*)(ws + 524288ULL);    // 512 KB
  unsigned short* xc   = (unsigned short*)(ws + 1048576ULL);   // 34 MB (reused as attn_out)
  unsigned short* qbuf = (unsigned short*)(ws + 35127296ULL);  // 34 MB
  unsigned short* xkv  = (unsigned short*)(ws + 69206016ULL);  // 34 MB
  unsigned short* kbuf = (unsigned short*)(ws + 103284736ULL); // 34 MB
  unsigned short* vt   = (unsigned short*)(ws + 137363456ULL); // 48 MB (token-padded V^T)
  unsigned short* up   = (unsigned short*)(ws + 187695104ULL); // 128 MB conv taps

  pack_w_k<<<2048, 256, 0, stream>>>(q_w, kv_w, p_w, sr_w, wq, wkv, wp, wsr);
  build_inputs_k<<<NWIN * NTOK, 256, 0, stream>>>(x, gt, xc, xkv, out);
  zero_k<<<12288, 256, 0, stream>>>((uint4*)vt, 50331648ULL / 16ULL);
  bilinear_k<<<262144, 256, 0, stream>>>(x, up);
  // q = xc @ q_w^T + q_b
  gemm_bias_k<2, 0><<<MROWS / 16, 256, 0, stream>>>(xc, wq, q_b, 256,
                                                    qbuf, nullptr, nullptr, nullptr);
  // conv + LN -> xkv tokens 1..64
  conv_ln_k<<<NWIN, 256, 0, stream>>>(up, wsr, sr_b, nw, nb, xkv);
  // kv = xkv @ kv_w^T + kv_b  -> k rows, transposed+padded v
  gemm_bias_k<4, 1><<<MROWS / 16, 256, 0, stream>>>(xkv, wkv, kv_b, 512,
                                                    nullptr, kbuf, vt, nullptr);
  // softmax attention (attn result reuses the xc region)
  attention_k<<<NWIN, 256, 0, stream>>>(qbuf, kbuf, vt, xc);
  // out projection + window scatter into d_out
  gemm_bias_k<2, 2><<<MROWS / 16, 256, 0, stream>>>(xc, wp, p_b, 256,
                                                    nullptr, nullptr, nullptr, out);
}
